// Decoder_70231305225195
// MI455X (gfx1250) — compile-verified, matrix operand order checked by reading, not run on hardware
//
#include <hip/hip_runtime.h>
#include <hip/hip_bf16.h>

// ---------------- problem constants ----------------
#define B_     2
#define S_     2048
#define HID_   1024
#define NH_    16
#define NKV_   4
#define HD_    64
#define E_     8
#define TOPK_  2
#define INTER_ 2816
#define WIN_   64
#define NREP_  (NH_ / NKV_)
#define NTOK_  (B_ * S_)        // 4096

typedef __bf16 bf16;
typedef __attribute__((ext_vector_type(16))) __bf16 v16bf;
typedef __attribute__((ext_vector_type(8)))  float  v8f;

typedef unsigned int u32x4 __attribute__((ext_vector_type(4)));
typedef int          i32x4 __attribute__((ext_vector_type(4)));
typedef int          i32x8 __attribute__((ext_vector_type(8)));

#if defined(__HIP_DEVICE_COMPILE__) && __has_builtin(__builtin_amdgcn_tensor_load_to_lds)
#define USE_TDM 1
#else
#define USE_TDM 0
#endif

// LDS tile geometry (fp32 tiles; padded row strides for bank spread)
#define AROW 34              // 32 K + 2 pad dwords
#define BROW 130             // 128 N + 2 pad dwords
#define ASZ  (128 * AROW)    // floats
#define BSZ  (32 * BROW)     // floats

#if USE_TDM
typedef __attribute__((address_space(3))) void lds_void;
__device__ __forceinline__ unsigned int lds_off(void* p)
{
    return (unsigned int)(size_t)(lds_void*)(void*)p;
}

// Issue one 2D TDM tile load: tile0 contiguous elems/row, tile1 rows,
// row stride strideElems (fp32 data), LDS padding via interval/amount codes.
__device__ __forceinline__ void tdm_load_2d(
    unsigned int ldsByteOff, const float* g,
    unsigned int tile0, unsigned int tile1, unsigned long long strideElems,
    unsigned int padIntCode, unsigned int padAmtCode)
{
    unsigned long long ga = (unsigned long long)(size_t)g;
    u32x4 g0;
    g0[0] = 1u;                                   // count=1, user descriptor
    g0[1] = ldsByteOff;                           // lds_addr
    g0[2] = (unsigned int)ga;                     // global_addr[31:0]
    g0[3] = ((unsigned int)(ga >> 32) & 0x01FFFFFFu) | (2u << 30); // addr[56:32] | type=2

    i32x8 g1;
    g1[0] = (int)((2u << 16)                      // data_size = 4B
                | (1u << 20)                      // pad_enable
                | (padIntCode << 22)              // pad_interval code
                | (padAmtCode << 25));            // pad_amount code
    g1[1] = (int)((tile0 & 0xFFFFu) << 16);       // tensor_dim0[15:0]
    g1[2] = (int)((tile0 >> 16) | ((tile1 & 0xFFFFu) << 16)); // td0 hi | tensor_dim1 lo
    g1[3] = (int)((tile1 >> 16) | (tile0 << 16)); // td1 hi | tile_dim0
    g1[4] = (int)(tile1 & 0xFFFFu);               // tile_dim1 (tile_dim2 = 0)
    g1[5] = (int)(unsigned int)strideElems;       // tensor_dim0_stride lo
    g1[6] = (int)(unsigned int)((strideElems >> 32) & 0xFFFFu);
    g1[7] = 0;

    i32x4 gz = {0, 0, 0, 0};
#if __clang_major__ >= 23
    i32x8 gz8 = {0, 0, 0, 0, 0, 0, 0, 0};
    __builtin_amdgcn_tensor_load_to_lds(g0, g1, gz, gz, gz8, 0);
#else
    __builtin_amdgcn_tensor_load_to_lds(g0, g1, gz, gz, 0);
#endif
}
#endif // USE_TDM

__device__ __forceinline__ bf16 tobf(float f) { return (bf16)f; }

// =====================================================================
// Generic WMMA GEMM: C[M,N] = A[M,K] @ B[K,N]  (fp32 in/out, bf16 WMMA)
// MODE 0: C = D;  MODE 1: C = D + R;  MODE 2: C = D + C (accumulate)
// Optional token compaction for MoE:
//   mapA   : gather A rows      (A row = mapA[row])
//   mapC   : scatter C rows     (C row = mapC[row]); rows >= *cntPtr skipped
//   cntPtr : real row count; blocks with bm >= padded count exit early
// M % 128 == 0, N % 128 == 0, K % 32 == 0 (true for all uses here).
// Block: 256 thr (8 waves). Each wave: 32x64 patch = 2x4 tiles of 16x16.
// TDM path: double-buffered tiles, TDM issued by wave 0, gated by
// s_wait_tensorcnt + workgroup barrier. Gathered A rows (mapA) go through
// vector loads; weights (B) always ride the TDM.
// =====================================================================
template <int MODE>
__global__ __launch_bounds__(256) void gemm_wmma(
    const float* __restrict__ A, const float* __restrict__ B,
    float* __restrict__ C, const float* __restrict__ R,
    int M, int N, int K,
    const int* __restrict__ mapA, const int* __restrict__ mapC,
    const int* __restrict__ cntPtr)
{
    __shared__ float sm[2 * (ASZ + BSZ)];

    const int tid  = threadIdx.x;
    const int lane = tid & 31;
    const int wid  = tid >> 5;
    const int bm   = blockIdx.y * 128;
    const int bn   = blockIdx.x * 128;
    const int wm   = (wid & 3) * 32;   // wave row offset in block
    const int wn   = (wid >> 2) * 64;  // wave col offset in block

    int creal = M;
    if (cntPtr) {
        creal = *cntPtr;
        int cp = (creal + 127) & ~127;
        if (bm >= cp) return;          // whole block exits together
    }

    v8f acc[2][4];
    for (int i = 0; i < 2; ++i)
        for (int j = 0; j < 4; ++j)
            acc[i][j] = (v8f){0.f,0.f,0.f,0.f,0.f,0.f,0.f,0.f};

    const int half = lane >> 4;     // K-half selector for 16-bit operands
    const int r16  = lane & 15;     // row (A) / col (B) within 16
    const int KT   = K / 32;

    // staging coordinates for the (possibly gathered) A tile
    const int stRow = tid >> 1;
    const int stKb  = (tid & 1) * 16;
    const int gRow  = mapA ? mapA[bm + stRow] : (bm + stRow);

#if USE_TDM
    const unsigned int baseOff = lds_off(sm);
    const bool tdmA = (mapA == nullptr);
    if (wid == 0) {
        if (tdmA)
            tdm_load_2d(baseOff, A + (size_t)bm * K, 32u, 128u,
                        (unsigned long long)K, 4u, 1u);
        tdm_load_2d(baseOff + (unsigned)(2 * ASZ) * 4u, B + bn, 128u, 32u,
                    (unsigned long long)N, 6u, 1u);
    }
#endif

    for (int kt = 0; kt < KT; ++kt) {
        const int cur = kt & 1;

#if USE_TDM
        if (!tdmA) {
            // gathered A rows staged through vector loads (double-buffered)
            const float* ga = A + (size_t)gRow * K + kt * 32 + stKb;
            float* sa = sm + cur * ASZ + stRow * AROW + stKb;
            for (int j = 0; j < 16; j += 4) {
                float4 f = *(const float4*)(ga + j);
                sa[j + 0] = f.x; sa[j + 1] = f.y; sa[j + 2] = f.z; sa[j + 3] = f.w;
            }
        }
        if (wid == 0) {
            if (kt + 1 < KT) {
                const int nxt = cur ^ 1;
                if (tdmA) {
                    tdm_load_2d(baseOff + (unsigned)(nxt * ASZ) * 4u,
                                A + (size_t)bm * K + (kt + 1) * 32, 32u, 128u,
                                (unsigned long long)K, 4u, 1u);
                    tdm_load_2d(baseOff + (unsigned)(2 * ASZ + nxt * BSZ) * 4u,
                                B + (size_t)((kt + 1) * 32) * N + bn, 128u, 32u,
                                (unsigned long long)N, 6u, 1u);
                    __builtin_amdgcn_s_wait_tensorcnt(2);  // current pair landed
                } else {
                    tdm_load_2d(baseOff + (unsigned)(2 * ASZ + nxt * BSZ) * 4u,
                                B + (size_t)((kt + 1) * 32) * N + bn, 128u, 32u,
                                (unsigned long long)N, 6u, 1u);
                    __builtin_amdgcn_s_wait_tensorcnt(1);  // current B landed
                }
            } else {
                __builtin_amdgcn_s_wait_tensorcnt(0);
            }
        }
        __syncthreads();
        const float* As = sm + cur * ASZ;
        const float* Bs = sm + 2 * ASZ + cur * BSZ;
#else
        // fallback: synchronous fp32 staging into buffer 0
        {
            const float* ga = A + (size_t)gRow * K + kt * 32 + stKb;
            float* sa = sm + stRow * AROW + stKb;
            for (int j = 0; j < 16; j += 4) {
                float4 f = *(const float4*)(ga + j);
                sa[j + 0] = f.x; sa[j + 1] = f.y; sa[j + 2] = f.z; sa[j + 3] = f.w;
            }
        }
        {
            int krow = tid >> 3, nb = (tid & 7) * 16;
            const float* gb = B + (size_t)(kt * 32 + krow) * N + bn + nb;
            float* sb = sm + 2 * ASZ + krow * BROW + nb;
            for (int j = 0; j < 16; j += 4) {
                float4 f = *(const float4*)(gb + j);
                sb[j + 0] = f.x; sb[j + 1] = f.y; sb[j + 2] = f.z; sb[j + 3] = f.w;
            }
        }
        __syncthreads();
        const float* As = sm;
        const float* Bs = sm + 2 * ASZ;
#endif

        // ---- build operand fragments in the 16-bit WMMA register layout
        //      lane: row = lane&15, K-half = lane>>4;
        //      VGPR v holds K pair at ((v<4)?2v:16+2(v-4)) + 8*half
        union Frag { v16bf v; bf16 e[16]; };
        Frag af[2], bfr[4];
        for (int mt = 0; mt < 2; ++mt) {
            const float* ar = As + (wm + mt * 16 + r16) * AROW;
            for (int v = 0; v < 8; ++v) {
                int kk = ((v < 4) ? 2 * v : 16 + 2 * (v - 4)) + 8 * half;
                af[mt].e[2 * v]     = tobf(ar[kk]);
                af[mt].e[2 * v + 1] = tobf(ar[kk + 1]);
            }
        }
        for (int nt = 0; nt < 4; ++nt) {
            const float* bc = Bs + (wn + nt * 16 + r16);
            for (int v = 0; v < 8; ++v) {
                int kk = ((v < 4) ? 2 * v : 16 + 2 * (v - 4)) + 8 * half;
                bfr[nt].e[2 * v]     = tobf(bc[kk * BROW]);
                bfr[nt].e[2 * v + 1] = tobf(bc[(kk + 1) * BROW]);
            }
        }
        for (int mt = 0; mt < 2; ++mt)
            for (int nt = 0; nt < 4; ++nt)
                acc[mt][nt] = __builtin_amdgcn_wmma_f32_16x16x32_bf16(
                    false, af[mt].v, false, bfr[nt].v,
                    (short)0, acc[mt][nt], false, false);
        __syncthreads();
    }

    // ---- epilogue: D layout lane -> col = lane&15, row = r + 8*(lane>>4)
    for (int mt = 0; mt < 2; ++mt) {
        for (int nt = 0; nt < 4; ++nt) {
            int col  = bn + wn + nt * 16 + r16;
            int rowb = bm + wm + mt * 16 + 8 * half;
            for (int r = 0; r < 8; ++r) {
                int lr = rowb + r;
                if (mapC && lr >= creal) continue;   // skip pad rows (no RMW race)
                int orow = mapC ? mapC[lr] : lr;
                size_t idx = (size_t)orow * N + col;
                float d = acc[mt][nt][r];
                if (MODE == 1) d += R[idx];
                if (MODE == 2) d += C[idx];
                C[idx] = d;
            }
        }
    }
}

// =====================================================================
// RMSNorm: one block per token
// =====================================================================
__global__ __launch_bounds__(256) void rmsnorm_kernel(
    const float* __restrict__ x, const float* __restrict__ w,
    float* __restrict__ out)
{
    __shared__ float red[8];
    int t = blockIdx.x;
    const float* xr = x + (size_t)t * HID_;
    float s = 0.f;
    for (int i = threadIdx.x; i < HID_; i += 256) { float v = xr[i]; s += v * v; }
    for (int o = 16; o > 0; o >>= 1) s += __shfl_xor(s, o, 32);
    if ((threadIdx.x & 31) == 0) red[threadIdx.x >> 5] = s;
    __syncthreads();
    if (threadIdx.x == 0) {
        float a = 0.f;
        for (int i = 0; i < 8; ++i) a += red[i];
        red[0] = a;
    }
    __syncthreads();
    float rs = rsqrtf(red[0] / (float)HID_ + 1e-8f);
    float* orow = out + (size_t)t * HID_;
    for (int i = threadIdx.x; i < HID_; i += 256) orow[i] = w[i] * xr[i] * rs;
}

// =====================================================================
// RoPE (in place), x laid out [tok][nh][HD], freqs [S][HD/2]
// =====================================================================
__global__ void rope_kernel(float* __restrict__ x,
                            const float* __restrict__ cosb,
                            const float* __restrict__ sinb, int nh)
{
    int idx = blockIdx.x * 256 + threadIdx.x;
    int total = NTOK_ * nh * (HD_ / 2);
    if (idx >= total) return;
    int j = idx & 31;
    int h = (idx >> 5) % nh;
    int t = idx / (32 * nh);
    int s = t & (S_ - 1);
    float c = cosb[s * 32 + j], sn = sinb[s * 32 + j];
    float* p = x + (size_t)t * nh * HD_ + h * HD_ + 2 * j;
    float xr = p[0], xi = p[1];
    p[0] = xr * c - xi * sn;
    p[1] = xr * sn + xi * c;
}

// =====================================================================
// Sliding-window causal attention (window 64, GQA). One wave per (b,h,q).
// K/V are only 4 MB -> L2-resident; online softmax, fp32.
// =====================================================================
__global__ __launch_bounds__(256) void attn_kernel(
    const float* __restrict__ q, const float* __restrict__ k,
    const float* __restrict__ v, float* __restrict__ out)
{
    int gw   = (blockIdx.x * 256 + threadIdx.x) >> 5;
    int lane = threadIdx.x & 31;
    int qpos = gw % S_;
    int h    = (gw / S_) % NH_;
    int b    = gw / (S_ * NH_);
    if (b >= B_) return;
    int kvh = h / NREP_;

    const float* qp = q + ((size_t)(b * S_ + qpos) * NH_ + h) * HD_;
    float q0 = qp[lane], q1 = qp[lane + 32];

    int kstart = qpos - WIN_; if (kstart < 0) kstart = 0;
    float m = -1e30f, l = 0.f, o0 = 0.f, o1 = 0.f;
    for (int kp = kstart; kp <= qpos; ++kp) {
        const float* kv = k + ((size_t)(b * S_ + kp) * NKV_ + kvh) * HD_;
        float sdot = q0 * kv[lane] + q1 * kv[lane + 32];
        for (int o = 16; o > 0; o >>= 1) sdot += __shfl_xor(sdot, o, 32);
        float sc = sdot * 0.125f;                 // 1/sqrt(64)
        float mn = fmaxf(m, sc);
        float rescale = __expf(m - mn);
        float p = __expf(sc - mn);
        const float* vv = v + ((size_t)(b * S_ + kp) * NKV_ + kvh) * HD_;
        l  = l * rescale + p;
        o0 = o0 * rescale + p * vv[lane];
        o1 = o1 * rescale + p * vv[lane + 32];
        m = mn;
    }
    float inv = 1.f / l;
    float* op = out + ((size_t)(b * S_ + qpos) * NH_ + h) * HD_;
    op[lane] = o0 * inv;
    op[lane + 32] = o1 * inv;
}

// =====================================================================
// Router: one wave per token; top-2 weights -> wall[tok][E], and
// atomic sums: sums[0..7]=sum probs, sums[8..15]=sum load counts
// =====================================================================
__global__ __launch_bounds__(256) void router_kernel(
    const float* __restrict__ x, const float* __restrict__ rw,
    float* __restrict__ wall, float* __restrict__ sums)
{
    int gw   = (blockIdx.x * 256 + threadIdx.x) >> 5;
    int lane = threadIdx.x & 31;
    if (gw >= NTOK_) return;
    const float* xr = x + (size_t)gw * HID_;
    float acc[E_];
    for (int e = 0; e < E_; ++e) acc[e] = 0.f;
    for (int i = lane; i < HID_; i += 32) {
        float xv = xr[i];
        const float* r = rw + (size_t)i * E_;
        for (int e = 0; e < E_; ++e) acc[e] += xv * r[e];
    }
    for (int e = 0; e < E_; ++e)
        for (int o = 16; o > 0; o >>= 1) acc[e] += __shfl_xor(acc[e], o, 32);
    if (lane == 0) {
        float mx = acc[0];
        for (int e = 1; e < E_; ++e) mx = fmaxf(mx, acc[e]);
        float pr[E_], den = 0.f;
        for (int e = 0; e < E_; ++e) { pr[e] = __expf(acc[e] - mx); den += pr[e]; }
        for (int e = 0; e < E_; ++e) pr[e] /= den;
        int i1 = 0;
        for (int e = 1; e < E_; ++e) if (pr[e] > pr[i1]) i1 = e;
        int i2 = -1;
        for (int e = 0; e < E_; ++e) {
            if (e == i1) continue;
            if (i2 < 0 || pr[e] > pr[i2]) i2 = e;
        }
        float ssum = pr[i1] + pr[i2] + 1e-9f;
        float w1 = pr[i1] / ssum, w2 = pr[i2] / ssum;
        for (int e = 0; e < E_; ++e)
            wall[(size_t)gw * E_ + e] = (e == i1) ? w1 : ((e == i2) ? w2 : 0.f);
        for (int e = 0; e < E_; ++e) atomicAdd(&sums[e], pr[e]);
        atomicAdd(&sums[E_ + i1], 1.f);
        atomicAdd(&sums[E_ + i2], 1.f);
    }
}

// =====================================================================
// MoE token compaction: per-expert token lists + weights
// =====================================================================
__global__ void zero_cnt_kernel(float* sums, int* cnt)
{
    if (threadIdx.x < 16) sums[threadIdx.x] = 0.f;
    if (threadIdx.x < E_) cnt[threadIdx.x] = 0;
}

__global__ void route_build_kernel(const float* __restrict__ wall,
                                   int* __restrict__ cnt,
                                   int* __restrict__ list,
                                   float* __restrict__ wgt)
{
    int t = blockIdx.x * 256 + threadIdx.x;
    if (t >= NTOK_) return;
    for (int e = 0; e < E_; ++e) {
        float w = wall[(size_t)t * E_ + e];
        if (w > 0.f) {
            int pos = atomicAdd(&cnt[e], 1);
            list[e * NTOK_ + pos] = t;
            wgt[e * NTOK_ + pos]  = w;
        }
    }
}

__global__ void route_pad_kernel(const int* __restrict__ cnt,
                                 int* __restrict__ list,
                                 float* __restrict__ wgt)
{
    for (int e = 0; e < E_; ++e) {
        int c  = cnt[e];
        int cp = (c + 127) & ~127;
        for (int j = c + threadIdx.x; j < cp; j += 256) {
            list[e * NTOK_ + j] = 0;
            wgt[e * NTOK_ + j]  = 0.f;
        }
    }
}

// H = silu(G) * U * wgt[row] over compacted rows (pad rows -> weight 0)
__global__ void silu_mul_c_kernel(float* __restrict__ G, const float* __restrict__ U,
                                  const float* __restrict__ wgt,
                                  const int* __restrict__ cntPtr)
{
    size_t idx = (size_t)blockIdx.x * 256 + threadIdx.x;
    if (idx >= (size_t)NTOK_ * INTER_) return;
    int row = (int)(idx / INTER_);
    int cp  = (*cntPtr + 127) & ~127;
    if (row >= cp) return;
    float g = G[idx];
    float sg = g / (1.f + __expf(-g));
    G[idx] = sg * U[idx] * wgt[row];
}

__global__ void copy_kernel(const float* __restrict__ src, float* __restrict__ dst, size_t n)
{
    size_t i = (size_t)blockIdx.x * 256 + threadIdx.x;
    if (i < n) dst[i] = src[i];
}

__global__ void loss_kernel(const float* __restrict__ sums, float* __restrict__ out)
{
    float a = 0.f;
    for (int e = 0; e < E_; ++e)
        a += (sums[e] / (float)NTOK_) * (sums[E_ + e] / (float)NTOK_);
    out[0] = (float)E_ * a * 0.01f;
}

// =====================================================================
// launch
// =====================================================================
extern "C" void kernel_launch(void* const* d_in, const int* in_sizes, int n_in,
                              void* d_out, int out_size, void* d_ws, size_t ws_size,
                              hipStream_t stream)
{
    const float* x        = (const float*)d_in[0];
    const float* fcos     = (const float*)d_in[1];
    const float* fsin     = (const float*)d_in[2];
    const float* wq       = (const float*)d_in[3];
    const float* wk       = (const float*)d_in[4];
    const float* wv       = (const float*)d_in[5];
    const float* wo       = (const float*)d_in[6];
    const float* router_w = (const float*)d_in[7];
    const float* gate_w   = (const float*)d_in[8];
    const float* up_w     = (const float*)d_in[9];
    const float* down_w   = (const float*)d_in[10];
    const float* n1       = (const float*)d_in[11];
    const float* n2       = (const float*)d_in[12];
    float* out = (float*)d_out;

    // workspace carve (floats)
    float* p   = (float*)d_ws;
    float* xn  = p; p += (size_t)NTOK_ * HID_;
    float* qb  = p; p += (size_t)NTOK_ * NH_ * HD_;
    float* kb  = p; p += (size_t)NTOK_ * NKV_ * HD_;
    float* vb  = p; p += (size_t)NTOK_ * NKV_ * HD_;
    float* atn = p; p += (size_t)NTOK_ * HID_;
    float* hb  = p; p += (size_t)NTOK_ * HID_;
    float* h2n = p; p += (size_t)NTOK_ * HID_;
    float* G   = p; p += (size_t)NTOK_ * INTER_;
    float* U   = p; p += (size_t)NTOK_ * INTER_;
    float* wal = p; p += (size_t)NTOK_ * E_;
    float* sms = p; p += 16;
    float* wgl = p; p += (size_t)E_ * NTOK_;     // compacted weights
    int*   cnt  = (int*)p; p += E_;              // per-expert counts
    int*   list = (int*)p; p += (size_t)E_ * NTOK_; // compacted token ids

    // 1) norm1
    rmsnorm_kernel<<<NTOK_, 256, 0, stream>>>(x, n1, xn);

    // 2) QKV projections (WMMA + TDM)
    dim3 gQ(HID_ / 128, NTOK_ / 128);
    dim3 gKV((NKV_ * HD_) / 128, NTOK_ / 128);
    gemm_wmma<0><<<gQ,  256, 0, stream>>>(xn, wq, qb, nullptr, NTOK_, NH_ * HD_,  HID_,
                                          nullptr, nullptr, nullptr);
    gemm_wmma<0><<<gKV, 256, 0, stream>>>(xn, wk, kb, nullptr, NTOK_, NKV_ * HD_, HID_,
                                          nullptr, nullptr, nullptr);
    gemm_wmma<0><<<gKV, 256, 0, stream>>>(xn, wv, vb, nullptr, NTOK_, NKV_ * HD_, HID_,
                                          nullptr, nullptr, nullptr);

    // 3) RoPE
    {
        int tq = NTOK_ * NH_ * (HD_ / 2);
        int tk = NTOK_ * NKV_ * (HD_ / 2);
        rope_kernel<<<(tq + 255) / 256, 256, 0, stream>>>(qb, fcos, fsin, NH_);
        rope_kernel<<<(tk + 255) / 256, 256, 0, stream>>>(kb, fcos, fsin, NKV_);
    }

    // 4) sliding-window attention
    attn_kernel<<<(B_ * NH_ * S_) / 8, 256, 0, stream>>>(qb, kb, vb, atn);

    // 5) output projection + residual (WMMA, fused residual)
    gemm_wmma<1><<<gQ, 256, 0, stream>>>(atn, wo, hb, x, NTOK_, HID_, NH_ * HD_,
                                         nullptr, nullptr, nullptr);

    // 6) norm2 + router + compaction
    rmsnorm_kernel<<<NTOK_, 256, 0, stream>>>(hb, n2, h2n);
    zero_cnt_kernel<<<1, 32, 0, stream>>>(sms, cnt);
    router_kernel<<<(NTOK_ + 7) / 8, 256, 0, stream>>>(h2n, router_w, wal, sms);
    route_build_kernel<<<NTOK_ / 256, 256, 0, stream>>>(wal, cnt, list, wgl);
    route_pad_kernel<<<1, 256, 0, stream>>>(cnt, list, wgl);

    // 7) out := residual; top-2 compacted experts accumulate into out
    copy_kernel<<<(int)(((size_t)NTOK_ * HID_ + 255) / 256), 256, 0, stream>>>(
        hb, out, (size_t)NTOK_ * HID_);

    dim3 gGU(INTER_ / 128, NTOK_ / 128);   // M dim = worst case, blocks early-exit
    dim3 gDN(HID_ / 128,  NTOK_ / 128);
    size_t guStride = (size_t)HID_ * INTER_;
    size_t dnStride = (size_t)INTER_ * HID_;
    int silBlocks = (int)(((size_t)NTOK_ * INTER_ + 255) / 256);
    for (int e = 0; e < E_; ++e) {
        const int* lst = list + (size_t)e * NTOK_;
        gemm_wmma<0><<<gGU, 256, 0, stream>>>(h2n, gate_w + e * guStride, G, nullptr,
                                              NTOK_, INTER_, HID_,
                                              lst, nullptr, cnt + e);
        gemm_wmma<0><<<gGU, 256, 0, stream>>>(h2n, up_w + e * guStride, U, nullptr,
                                              NTOK_, INTER_, HID_,
                                              lst, nullptr, cnt + e);
        silu_mul_c_kernel<<<silBlocks, 256, 0, stream>>>(G, U, wgl + (size_t)e * NTOK_,
                                                         cnt + e);
        gemm_wmma<2><<<gDN, 256, 0, stream>>>(G, down_w + e * dnStride, out, nullptr,
                                              NTOK_, HID_, INTER_,
                                              nullptr, lst, cnt + e);
    }

    // 8) router loss scalar appended after the [B,S,HID] output
    loss_kernel<<<1, 1, 0, stream>>>(sms, out + (size_t)NTOK_ * HID_);
}
